// Attn_64665027609188
// MI455X (gfx1250) — matrix operand order repaired and checked by reading, not compiled, not run on hardware
//
#include <hip/hip_runtime.h>
#include <math.h>

// ---------------------------------------------------------------------------
// Types for CDNA5 WMMA (wave32): D(16x16 f32) = A(16x32 bf16) x B(32x16 bf16) + C
// ---------------------------------------------------------------------------
typedef __bf16 bf16_t;
typedef __attribute__((ext_vector_type(16))) __bf16 v16bf;
typedef __attribute__((ext_vector_type(8)))  __bf16 v8bf;
typedef __attribute__((ext_vector_type(4)))  __bf16 v4bf;
typedef __attribute__((ext_vector_type(8)))  float  v8f;

#define WMMA_BF16(a, b, c) \
    __builtin_amdgcn_wmma_f32_16x16x32_bf16(false, (a), false, (b), (short)0, (c), false, false)

// Problem constants (match reference).
#define BATCH 2
#define SEQ   2048
#define DIM   4096
#define NH    32
#define NKV   8
#define HD    128
#define MROWS (BATCH * SEQ)   // 4096

// ---------------------------------------------------------------------------
// DPP16 butterfly reductions within each 16-lane row (VALU only; avoids the
// ds_bpermute + s_wait_dscnt path of __shfl_xor).
// ---------------------------------------------------------------------------
__device__ __forceinline__ float dpp_mov(float v, const int ctrl) {
    int r;
    switch (ctrl) {  // ctrl must be an immediate per builtin; constant-folded
        case 0xB1:  r = __builtin_amdgcn_update_dpp(0, __builtin_bit_cast(int, v), 0xB1,  0xF, 0xF, true); break;
        case 0x4E:  r = __builtin_amdgcn_update_dpp(0, __builtin_bit_cast(int, v), 0x4E,  0xF, 0xF, true); break;
        case 0x141: r = __builtin_amdgcn_update_dpp(0, __builtin_bit_cast(int, v), 0x141, 0xF, 0xF, true); break;
        default:    r = __builtin_amdgcn_update_dpp(0, __builtin_bit_cast(int, v), 0x140, 0xF, 0xF, true); break;
    }
    return __builtin_bit_cast(float, r);
}
__device__ __forceinline__ float row_max16(float x) {
    x = fmaxf(x, dpp_mov(x, 0xB1));   // xor 1
    x = fmaxf(x, dpp_mov(x, 0x4E));   // xor 2
    x = fmaxf(x, dpp_mov(x, 0x141));  // xor 7 (combine quads within 8)
    x = fmaxf(x, dpp_mov(x, 0x140));  // xor 15 (combine 8-groups within 16)
    return x;
}
__device__ __forceinline__ float row_sum16(float x) {
    x += dpp_mov(x, 0xB1);
    x += dpp_mov(x, 0x4E);
    x += dpp_mov(x, 0x141);
    x += dpp_mov(x, 0x140);
    return x;
}

// ---------------------------------------------------------------------------
// fp32 -> bf16 conversion (4 elements / thread)
// ---------------------------------------------------------------------------
__global__ void cvt_f32_bf16(const float* __restrict__ in, bf16_t* __restrict__ out, long n4) {
    long i = (long)blockIdx.x * blockDim.x + threadIdx.x;
    if (i >= n4) return;
    float4 f = ((const float4*)in)[i];
    v4bf o;
    o[0] = (bf16_t)f.x; o[1] = (bf16_t)f.y; o[2] = (bf16_t)f.z; o[3] = (bf16_t)f.w;
    ((v4bf*)out)[i] = o;
}

// ---------------------------------------------------------------------------
// In-place RoPE on bf16 tensor laid out (B, S, nheads, HD).
// ---------------------------------------------------------------------------
__global__ void rope_inplace_bf16(bf16_t* __restrict__ q,
                                  const float* __restrict__ cosT,
                                  const float* __restrict__ sinT,
                                  int nheads, long total) {
    long i = (long)blockIdx.x * blockDim.x + threadIdx.x;
    if (i >= total) return;
    int  j   = (int)(i & 63);           // HD/2 == 64
    long bsh = i >> 6;                  // flattened (b, s, h)
    int  s   = (int)((bsh / nheads) % SEQ);
    float c  = cosT[s * 64 + j];
    float sn = sinT[s * 64 + j];
    long base = bsh * HD + 2 * j;
    float xe = (float)q[base];
    float xo = (float)q[base + 1];
    q[base]     = (bf16_t)(xe * c - xo * sn);
    q[base + 1] = (bf16_t)(xe * sn + xo * c);
}

// ---------------------------------------------------------------------------
// Tiled bf16 GEMM, double-buffered LDS (one barrier per K-step).
// C(MxN, OutT) = A(MxK bf16 row-major) @ B(KxN bf16 row-major).
// Block tile 128x128, 256 threads = 8 waves (2x4), wave tile 64x32 = 4x2 WMMA.
// ---------------------------------------------------------------------------
#define GLDA 40
#define GLDB 40

template <typename OutT>
__global__ __launch_bounds__(256) void gemm_bf16(const bf16_t* __restrict__ A,
                                                 const bf16_t* __restrict__ B,
                                                 OutT* __restrict__ C,
                                                 int M, int N, int K) {
    __shared__ bf16_t Alds[2][128 * GLDA];   // [buf][row][k]
    __shared__ bf16_t Blds[2][128 * GLDB];   // [buf][n][k] (transposed)

    const int tid  = threadIdx.x;
    const int lane = tid & 31;
    const int wid  = tid >> 5;
    const int lh   = lane & 15;
    const int hi   = lane >> 4;              // 0/1 half-wave

    const int m0 = blockIdx.y * 128;
    const int n0 = blockIdx.x * 128;
    const int wm = (wid >> 2) * 64;          // wave M offset: 0/64
    const int wn = (wid & 3) * 32;           // wave N offset: 0/32/64/96

    // cooperative-load coordinates
    const int arow = tid >> 2;               // 0..63
    const int acol = (tid & 3) * 8;          // 0,8,16,24
    const int bn   = (tid & 31) * 4;         // 0..124
    const int bk   = tid >> 5;               // 0..7

    auto stage = [&](int kk, int bsel) {
#pragma unroll
        for (int p = 0; p < 2; p++) {
            int r = arow + p * 64;
            *(v8bf*)&Alds[bsel][r * GLDA + acol] =
                *(const v8bf*)&A[(size_t)(m0 + r) * K + kk + acol];
        }
#pragma unroll
        for (int p = 0; p < 4; p++) {
            int k = bk + p * 8;
            v4bf b = *(const v4bf*)&B[(size_t)(kk + k) * N + n0 + bn];
            Blds[bsel][(bn + 0) * GLDB + k] = b[0];
            Blds[bsel][(bn + 1) * GLDB + k] = b[1];
            Blds[bsel][(bn + 2) * GLDB + k] = b[2];
            Blds[bsel][(bn + 3) * GLDB + k] = b[3];
        }
    };

    v8f acc[4][2];
#pragma unroll
    for (int mt = 0; mt < 4; mt++)
#pragma unroll
        for (int nt = 0; nt < 2; nt++) acc[mt][nt] = {};

    stage(0, 0);
    __syncthreads();

    int buf = 0;
    for (int kk = 0; kk < K; kk += 32) {
        if (kk + 32 < K) stage(kk + 32, buf ^ 1);   // prefetch into other buffer

        // A fragments: lane<16 holds K 0-7 & 16-23 of row lh (ISA 7.12.2)
        v16bf afrag[4];
#pragma unroll
        for (int mt = 0; mt < 4; mt++) {
            int row = wm + mt * 16 + lh;
            int kb  = hi * 8;
            v8bf lo = *(const v8bf*)&Alds[buf][row * GLDA + kb];
            v8bf hv = *(const v8bf*)&Alds[buf][row * GLDA + kb + 16];
            v16bf a;
#pragma unroll
            for (int i = 0; i < 8; i++) { a[i] = lo[i]; a[8 + i] = hv[i]; }
            afrag[mt] = a;
        }
        // B fragments: lane<16 holds K 0-7 & 16-23 of column lh
        v16bf bfrag[2];
#pragma unroll
        for (int nt = 0; nt < 2; nt++) {
            int col = wn + nt * 16 + lh;
            int kb  = hi * 8;
            v8bf lo = *(const v8bf*)&Blds[buf][col * GLDB + kb];
            v8bf hv = *(const v8bf*)&Blds[buf][col * GLDB + kb + 16];
            v16bf b;
#pragma unroll
            for (int i = 0; i < 8; i++) { b[i] = lo[i]; b[8 + i] = hv[i]; }
            bfrag[nt] = b;
        }
#pragma unroll
        for (int mt = 0; mt < 4; mt++)
#pragma unroll
            for (int nt = 0; nt < 2; nt++)
                acc[mt][nt] = WMMA_BF16(afrag[mt], bfrag[nt], acc[mt][nt]);

        __syncthreads();      // reads of buf done, writes to buf^1 done
        buf ^= 1;
    }

    // D layout: lane = column, VGPR r = row r + 8*hi
#pragma unroll
    for (int mt = 0; mt < 4; mt++)
#pragma unroll
        for (int nt = 0; nt < 2; nt++) {
            int col = n0 + wn + nt * 16 + lh;
#pragma unroll
            for (int r = 0; r < 8; r++) {
                int row = m0 + wm + mt * 16 + r + 8 * hi;
                C[(size_t)row * N + col] = (OutT)acc[mt][nt][r];
            }
        }
}

// ---------------------------------------------------------------------------
// Flash attention (causal, GQA nrep=4). Grid: (S/64, B*NH), 128 threads.
// Each wave owns 16 query rows; block processes 64 queries x full HD=128.
// K staged in LDS as [key][hd] (score B-fragments), V transposed [hd][key]
// (PV B-fragments); both double-buffered. All B-fragments of a WMMA chain are
// preloaded into distinct registers so DS loads batch and drain with partial
// s_wait_dscnt instead of serializing load->wait(0)->wmma.
// ---------------------------------------------------------------------------
#define ALKH 136   // K lds stride: 128 + 8 pad (bf16 units, 16B-aligned rows)
#define ALKV 40    // V/P lds stride: 32 + 8 pad

__global__ __launch_bounds__(128) void attn_fwd(const bf16_t* __restrict__ Q,
                                                const bf16_t* __restrict__ Kt,
                                                const bf16_t* __restrict__ V,
                                                bf16_t* __restrict__ O) {
    __shared__ bf16_t Klds[2][32 * ALKH];
    __shared__ bf16_t Vlds[2][HD * ALKV];
    __shared__ bf16_t Plds[4 * 16 * ALKV];

    const float scale = 0.08838834764831845f;   // 1/sqrt(HD)

    const int tid  = threadIdx.x;
    const int lane = tid & 31;
    const int wid  = tid >> 5;
    const int lh   = lane & 15;
    const int hi   = lane >> 4;

    const int qb  = blockIdx.x;                 // 64-query block index
    const int bh  = blockIdx.y;
    const int b   = bh / NH;
    const int h   = bh % NH;
    const int hkv = h / (NH / NKV);
    const int qw  = qb * 64 + wid * 16;         // wave's first query row

    // Q A-fragments for the 4 K-steps over HD (held for the whole loop)
    v16bf aq[4];
    {
        const bf16_t* qp = Q + ((size_t)(b * SEQ + qw + lh) * NH + h) * HD;
#pragma unroll
        for (int ks = 0; ks < 4; ks++) {
            int base = ks * 32 + hi * 8;
            v8bf lo = *(const v8bf*)&qp[base];
            v8bf hv = *(const v8bf*)&qp[base + 16];
            v16bf a;
#pragma unroll
            for (int i = 0; i < 8; i++) { a[i] = lo[i]; a[8 + i] = hv[i]; }
            aq[ks] = a;
        }
    }

    v8f o_acc[8];
#pragma unroll
    for (int t = 0; t < 8; t++) o_acc[t] = {};
    float m_s[8], l_s[8];
#pragma unroll
    for (int r = 0; r < 8; r++) { m_s[r] = -1.0e30f; l_s[r] = 0.0f; }

    const int lkey = tid >> 2;                  // 0..31
    const int lhd  = (tid & 3) * 32;            // 0,32,64,96
    const int kend = (qb + 1) * 64;             // causal: skip fully-masked blocks

    auto stage = [&](int kb, int bsel) {
        const size_t base = ((size_t)(b * SEQ + kb + lkey) * NKV + hkv) * HD + lhd;
        const bf16_t* kp = Kt + base;
        const bf16_t* vp = V + base;
#pragma unroll
        for (int c = 0; c < 4; c++)
            *(v8bf*)&Klds[bsel][lkey * ALKH + lhd + c * 8] = *(const v8bf*)&kp[c * 8];
#pragma unroll
        for (int i = 0; i < 32; i++)
            Vlds[bsel][(lhd + i) * ALKV + lkey] = vp[i];
    };

    stage(0, 0);
    __syncthreads();

    int buf = 0;
    for (int kb = 0; kb < kend; kb += 32) {
        if (kb + 32 < kend) stage(kb + 32, buf ^ 1);   // prefetch next K/V block

        // ---- scores: preload all 8 B-fragments, then 8 WMMAs ----
        v16bf bfr[4][2];
#pragma unroll
        for (int ks = 0; ks < 4; ks++)
#pragma unroll
            for (int nt = 0; nt < 2; nt++) {
                int key = nt * 16 + lh;
                int kbs = ks * 32 + hi * 8;
                v8bf lo = *(const v8bf*)&Klds[buf][key * ALKH + kbs];
                v8bf hv = *(const v8bf*)&Klds[buf][key * ALKH + kbs + 16];
                v16bf bfv;
#pragma unroll
                for (int i = 0; i < 8; i++) { bfv[i] = lo[i]; bfv[8 + i] = hv[i]; }
                bfr[ks][nt] = bfv;
            }
        v8f sc[2];
        sc[0] = {}; sc[1] = {};
#pragma unroll
        for (int ks = 0; ks < 4; ks++)
#pragma unroll
            for (int nt = 0; nt < 2; nt++)
                sc[nt] = WMMA_BF16(aq[ks], bfr[ks][nt], sc[nt]);

        // ---- scale + causal mask + online softmax (DPP row reductions) ----
        float pm[8];
#pragma unroll
        for (int r = 0; r < 8; r++) pm[r] = -1.0e30f;
#pragma unroll
        for (int nt = 0; nt < 2; nt++) {
            int kidx = kb + nt * 16 + lh;
#pragma unroll
            for (int r = 0; r < 8; r++) {
                int qrow = qw + r + 8 * hi;
                float sv = sc[nt][r] * scale;
                if (kidx > qrow) sv += -1.0e9f;
                sc[nt][r] = sv;
                pm[r] = fmaxf(pm[r], sv);
            }
        }
#pragma unroll
        for (int r = 0; r < 8; r++) pm[r] = row_max16(pm[r]);

        float alpha[8], rsum[8];
#pragma unroll
        for (int r = 0; r < 8; r++) {
            float mn = fmaxf(m_s[r], pm[r]);
            alpha[r] = __expf(m_s[r] - mn);
            m_s[r]   = mn;
            rsum[r]  = 0.0f;
        }
#pragma unroll
        for (int nt = 0; nt < 2; nt++)
#pragma unroll
            for (int r = 0; r < 8; r++) {
                float p = __expf(sc[nt][r] - m_s[r]);
                sc[nt][r] = p;
                rsum[r] += p;
            }
#pragma unroll
        for (int r = 0; r < 8; r++) rsum[r] = row_sum16(rsum[r]);
#pragma unroll
        for (int r = 0; r < 8; r++) l_s[r] = l_s[r] * alpha[r] + rsum[r];
#pragma unroll
        for (int t = 0; t < 8; t++)
#pragma unroll
            for (int r = 0; r < 8; r++) o_acc[t][r] *= alpha[r];

        // ---- P (D-layout) -> wave-private LDS -> A-fragment ----
        bf16_t* pw = &Plds[wid * 16 * ALKV];
#pragma unroll
        for (int nt = 0; nt < 2; nt++)
#pragma unroll
            for (int r = 0; r < 8; r++)
                pw[(r + 8 * hi) * ALKV + nt * 16 + lh] = (bf16_t)sc[nt][r];
        v16bf pa;
        {
            int kbq = hi * 8;
            v8bf lo = *(const v8bf*)&pw[lh * ALKV + kbq];
            v8bf hv = *(const v8bf*)&pw[lh * ALKV + kbq + 16];
#pragma unroll
            for (int i = 0; i < 8; i++) { pa[i] = lo[i]; pa[8 + i] = hv[i]; }
        }

        // ---- O += P(16x32) @ V(32x128): preload 8 V fragments, then WMMAs ----
        v16bf vfr[8];
#pragma unroll
        for (int nt = 0; nt < 8; nt++) {
            int hdc = nt * 16 + lh;
            int kb2 = hi * 8;
            v8bf lo = *(const v8bf*)&Vlds[buf][hdc * ALKV + kb2];
            v8bf hv = *(const v8bf*)&Vlds[buf][hdc * ALKV + kb2 + 16];
            v16bf bv;
#pragma unroll
            for (int i = 0; i < 8; i++) { bv[i] = lo[i]; bv[8 + i] = hv[i]; }
            vfr[nt] = bv;
        }
#pragma unroll
        for (int nt = 0; nt < 8; nt++)
            o_acc[nt] = WMMA_BF16(pa, vfr[nt], o_acc[nt]);

        __syncthreads();
        buf ^= 1;
    }

    // ---- normalize and emit bf16 (B,S,NH,HD) for the output projection ----
#pragma unroll
    for (int nt = 0; nt < 8; nt++)
#pragma unroll
        for (int r = 0; r < 8; r++) {
            int qrow = qw + r + 8 * hi;
            float val = o_acc[nt][r] / l_s[r];
            O[((size_t)(b * SEQ + qrow) * NH + h) * HD + nt * 16 + lh] = (bf16_t)val;
        }
}

// ---------------------------------------------------------------------------
// Host orchestration. Inputs: x, wq, wk, wv, wo, freqs_cos, freqs_sin, mask,
// start_pos (mask ignored — causal computed inline; start_pos == 0).
// Workspace (~192 MB): bf16 copies of x + weights, Q/K/V (bf16, roped in
// place), attention output (bf16). Final GEMM writes fp32 to d_out.
// ---------------------------------------------------------------------------
extern "C" void kernel_launch(void* const* d_in, const int* in_sizes, int n_in,
                              void* d_out, int out_size, void* d_ws, size_t ws_size,
                              hipStream_t stream) {
    (void)in_sizes; (void)n_in; (void)out_size; (void)ws_size;
    const float* x  = (const float*)d_in[0];
    const float* wq = (const float*)d_in[1];
    const float* wk = (const float*)d_in[2];
    const float* wv = (const float*)d_in[3];
    const float* wo = (const float*)d_in[4];
    const float* fc = (const float*)d_in[5];
    const float* fs = (const float*)d_in[6];

    const long nX  = (long)MROWS * DIM;        // 16,777,216
    const long nWq = (long)DIM * (NH * HD);    // 16,777,216
    const long nWk = (long)DIM * (NKV * HD);   //  4,194,304
    const long nO  = (long)MROWS * (NH * HD);  // 16,777,216 (== out elems)
    const long nK  = (long)MROWS * (NKV * HD); //  4,194,304

    char* ws = (char*)d_ws;
    size_t off = 0;
    auto take = [&](size_t elems) {
        bf16_t* p = (bf16_t*)(ws + off);
        off += (elems * sizeof(bf16_t) + 255) & ~(size_t)255;
        return p;
    };
    bf16_t* xb  = take(nX);
    bf16_t* wqb = take(nWq);
    bf16_t* wkb = take(nWk);
    bf16_t* wvb = take(nWk);
    bf16_t* wob = take(nWq);
    bf16_t* qb  = take(nO);
    bf16_t* kbf = take(nK);
    bf16_t* vbf = take(nK);
    bf16_t* ab  = take(nO);

    auto cvt = [&](const float* in, bf16_t* outp, long n) {
        long n4 = n / 4;
        cvt_f32_bf16<<<dim3((unsigned)((n4 + 255) / 256)), 256, 0, stream>>>(in, outp, n4);
    };
    cvt(x, xb, nX);
    cvt(wq, wqb, nWq);
    cvt(wk, wkb, nWk);
    cvt(wv, wvb, nWk);
    cvt(wo, wob, nWq);

    // QKV projections (bf16 out)
    gemm_bf16<bf16_t><<<dim3((NH * HD) / 128, MROWS / 128), 256, 0, stream>>>(
        xb, wqb, qb, MROWS, NH * HD, DIM);
    gemm_bf16<bf16_t><<<dim3((NKV * HD) / 128, MROWS / 128), 256, 0, stream>>>(
        xb, wkb, kbf, MROWS, NKV * HD, DIM);
    gemm_bf16<bf16_t><<<dim3((NKV * HD) / 128, MROWS / 128), 256, 0, stream>>>(
        xb, wvb, vbf, MROWS, NKV * HD, DIM);

    // RoPE (in place, bf16)
    {
        long tq = (long)BATCH * SEQ * NH * (HD / 2);
        long tk = (long)BATCH * SEQ * NKV * (HD / 2);
        rope_inplace_bf16<<<dim3((unsigned)((tq + 255) / 256)), 256, 0, stream>>>(qb, fc, fs, NH, tq);
        rope_inplace_bf16<<<dim3((unsigned)((tk + 255) / 256)), 256, 0, stream>>>(kbf, fc, fs, NKV, tk);
    }

    // Flash attention
    attn_fwd<<<dim3(SEQ / 64, BATCH * NH), 128, 0, stream>>>(qb, kbf, vbf, ab);

    // Output projection (fp32 straight into d_out)
    gemm_bf16<float><<<dim3(DIM / 128, MROWS / 128), 256, 0, stream>>>(
        ab, wob, (float*)d_out, MROWS, DIM, DIM);
}
